// ParticleNet_9715216023598
// MI455X (gfx1250) — matrix operand order repaired and checked by reading, not compiled, not run on hardware
//
#include <hip/hip_runtime.h>
#include <hip/hip_bf16.h>

typedef __attribute__((ext_vector_type(16))) _Float16 v16h;
typedef __attribute__((ext_vector_type(8)))  float    v8f;

#define B_   512
#define P_   128
#define K_   16
#define C0_  64
#define C1_  128
#define FC_  256
#define EPS_ 1e-5f

__device__ __forceinline__ v8f wmma_f16f32(v16h a, v16h b, v8f c) {
  // D = A(16x32 f16) * B(32x16 f16) + C(16x16 f32)
  return __builtin_amdgcn_wmma_f32_16x16x32_f16(false, a, false, b, (short)0, c,
                                                false, false);
}

// CDNA5 async global->LDS copy (16B per lane), tracked by ASYNCcnt.
// lds_off is the byte offset inside the block's dynamic-LDS allocation.
__device__ __forceinline__ void async_ld_b128(unsigned lds_off, const void* g) {
  asm volatile("global_load_async_to_lds_b128 %0, %1, off"
               :: "v"(lds_off), "v"(g) : "memory");
}
__device__ __forceinline__ void wait_async0() {
  asm volatile("s_wait_asynccnt 0x0" ::: "memory");
}

// ---------------- sorted 16-way insertion (fully unrolled -> registers) -------
__device__ __forceinline__ void insert16(float d2, int q, float dl[16], int il[16]) {
  if (d2 < dl[15]) {
#pragma unroll
    for (int s = 15; s > 0; --s) {
      if (d2 < dl[s - 1])      { dl[s] = dl[s - 1]; il[s] = il[s - 1]; }
      else if (d2 < dl[s])     { dl[s] = d2;        il[s] = q; }
    }
    if (d2 < dl[0])            { dl[0] = d2;        il[0] = q; }
  }
}

// ---------------- kernel 0: zero stats accumulators ---------------------------
__global__ void init_stats_kernel(float* stats0, float* stats1) {
  int t = threadIdx.x;
  if (t < 128) stats0[t] = 0.f;
  if (t < 256) stats1[t] = 0.f;
}

// ---------------- kernel 1: kNN on first 2 coords -----------------------------
__global__ __launch_bounds__(128) void knn0_kernel(const float* __restrict__ x,
                                                   int* __restrict__ idx0) {
  __shared__ float cx[P_], cy[P_];
  const int b = blockIdx.x, p = threadIdx.x;
  const float* xp = x + ((size_t)b * P_ + p) * 4;
  cx[p] = xp[0]; cy[p] = xp[1];
  __syncthreads();
  const float px = cx[p], py = cy[p];
  float dl[16]; int il[16];
#pragma unroll
  for (int s = 0; s < 16; ++s) { dl[s] = 3.0e38f; il[s] = 0; }
  for (int q = 0; q < P_; ++q) {
    float dx = px - cx[q], dy = py - cy[q];
    float d2 = dx * dx + dy * dy;
    if (q == p) d2 += 1e9f;
    insert16(d2, q, dl, il);
  }
  int* op = idx0 + ((size_t)b * P_ + p) * K_;
#pragma unroll
  for (int s = 0; s < 16; ++s) op[s] = il[s];
}

// ---------------- kernel 2: EdgeConv0 pass1 (BN stats) ------------------------
__global__ __launch_bounds__(256) void ec0_pass1_kernel(
    const float* __restrict__ x, const int* __restrict__ idx0,
    const float* __restrict__ w0, const float* __restrict__ b0,
    float* __restrict__ stats0) {
  __shared__ float sred[2 * C0_];
  const int b = blockIdx.x, tid = threadIdx.x;
  const int c = tid & 63, pg = tid >> 6;
  float wc[8];
#pragma unroll
  for (int f = 0; f < 8; ++f) wc[f] = w0[f * C0_ + c];
  const float bias = b0[c];
  float s = 0.f, q = 0.f;
  for (int p = pg; p < P_; p += 4) {
    const float* xi = x + ((size_t)b * P_ + p) * 4;
    if (p + 4 < P_) __builtin_prefetch(xi + 16, 0, 1);   // global_prefetch_b8
    const float mi = bias + xi[0] * wc[0] + xi[1] * wc[1] + xi[2] * wc[2] + xi[3] * wc[3];
    const int* ip = idx0 + ((size_t)b * P_ + p) * K_;
    for (int e = 0; e < K_; ++e) {
      const float* xj = x + ((size_t)b * P_ + ip[e]) * 4;
      float m = mi + xj[0] * wc[4] + xj[1] * wc[5] + xj[2] * wc[6] + xj[3] * wc[7];
      s += m; q += m * m;
    }
  }
  if (tid < 128) sred[tid] = 0.f;
  __syncthreads();
  atomicAdd(&sred[c], s);
  atomicAdd(&sred[C0_ + c], q);
  __syncthreads();
  if (tid < 128) atomicAdd(&stats0[tid], sred[tid]);
}

// ---------------- kernel 3: EdgeConv0 pass2 (BN+ReLU, act0, h0->f16) ----------
__global__ __launch_bounds__(64) void ec0_pass2_kernel(
    const float* __restrict__ x, const int* __restrict__ idx0,
    const float* __restrict__ w0, const float* __restrict__ b0,
    const float* __restrict__ g0, const float* __restrict__ beta0,
    const float* __restrict__ stats0,
    float* __restrict__ act0, _Float16* __restrict__ h0f16) {
  const int bp = blockIdx.x;        // b*128+p
  const int c  = threadIdx.x;       // channel
  float wc[8];
#pragma unroll
  for (int f = 0; f < 8; ++f) wc[f] = w0[f * C0_ + c];
  const float invN = 1.f / (float)(B_ * P_ * K_);
  const float mean = stats0[c] * invN;
  const float var  = stats0[C0_ + c] * invN - mean * mean;
  const float sc   = rsqrtf(var + EPS_) * g0[c];
  const float sh   = beta0[c] - mean * sc;
  const float* xi = x + (size_t)bp * 4;
  const float mi = b0[c] + xi[0] * wc[0] + xi[1] * wc[1] + xi[2] * wc[2] + xi[3] * wc[3];
  const int* ip = idx0 + (size_t)bp * K_;
  const size_t bbase = (size_t)(bp >> 7) << 7;
  float* ap = act0 + (size_t)bp * K_ * C0_ + c;
  float hs = 0.f;
  for (int e = 0; e < K_; ++e) {
    const float* xj = x + (bbase + (size_t)ip[e]) * 4;
    float m = mi + xj[0] * wc[4] + xj[1] * wc[5] + xj[2] * wc[6] + xj[3] * wc[7];
    float v = fmaxf(m * sc + sh, 0.f);
    ap[(size_t)e * C0_] = v;
    hs += v;
  }
  h0f16[(size_t)bp * C0_ + c] = (_Float16)(hs * (1.f / 16.f));
}

// ---------------- kernel 4: kNN on h0 via WMMA Gram matrix --------------------
#define HP_ 72                      // h LDS pitch (halves) -> 144B rows
#define HS_BYTES_ (P_ * HP_ * 2)    // 18432
#define GP_ 132                     // gram LDS pitch (floats)
__global__ __launch_bounds__(256) void knn1_kernel(const _Float16* __restrict__ h0,
                                                   int* __restrict__ idx1) {
  extern __shared__ char smem[];
  _Float16* Hs   = (_Float16*)smem;                          // 128 x HP_
  float*    nrm  = (float*)(smem + HS_BYTES_);               // 128
  float*    G    = (float*)(smem + HS_BYTES_ + 512);         // 128 x GP_
  const int b = blockIdx.x, tid = threadIdx.x;
  const _Float16* hb = h0 + (size_t)b * P_ * C0_;
  // async stage h0 tile: 128 rows x 128B, 8 chunks of 16B per row (pitch 144B)
  for (int i = tid; i < 1024; i += 256) {
    const int row = i >> 3, ch = i & 7;
    async_ld_b128((unsigned)(row * (HP_ * 2) + ch * 16), hb + row * C0_ + ch * 8);
  }
  wait_async0();
  __syncthreads();
  if (tid < P_) {
    float n = 0.f;
    for (int f = 0; f < C0_; ++f) { float v = (float)Hs[tid * HP_ + f]; n += v * v; }
    nrm[tid] = n;
  }
  __syncthreads();
  const int wave = tid >> 5, lane = tid & 31;
  const int hi = (lane >= 16) ? 1 : 0, m0 = lane & 15;
  // G = H * H^T : wave owns row-block, loops over 8 col-blocks, K=64 in 2 steps
  for (int jt = 0; jt < 8; ++jt) {
    v8f acc = {};
    for (int ks = 0; ks < C0_; ks += 32) {
      v16h a, bf;
      const _Float16* ar = &Hs[(16 * wave + m0) * HP_];
      const int ab = ks + hi * 8;
#pragma unroll
      for (int e = 0; e < 8; ++e) { a[e] = ar[ab + e]; a[8 + e] = ar[ab + 16 + e]; }
      const _Float16* br = &Hs[(16 * jt + m0) * HP_];
      const int bb = ks + hi * 8;
#pragma unroll
      for (int e = 0; e < 8; ++e) { bf[e] = br[bb + e]; bf[8 + e] = br[bb + 16 + e]; }
      acc = wmma_f16f32(a, bf, acc);
    }
#pragma unroll
    for (int r = 0; r < 8; ++r)
      G[(16 * wave + r + hi * 8) * GP_ + 16 * jt + m0] = acc[r];
  }
  __syncthreads();
  if (tid < P_) {
    const float ni = nrm[tid];
    float dl[16]; int il[16];
#pragma unroll
    for (int s = 0; s < 16; ++s) { dl[s] = 3.0e38f; il[s] = 0; }
    for (int q = 0; q < P_; ++q) {
      float d2 = ni + nrm[q] - 2.f * G[tid * GP_ + q];
      if (q == tid) d2 += 1e9f;
      insert16(d2, q, dl, il);
    }
    int* op = idx1 + ((size_t)b * P_ + tid) * K_;
#pragma unroll
    for (int s = 0; s < 16; ++s) op[s] = il[s];
  }
}

// ---------------- kernels 5/6: EdgeConv1 via WMMA (pass0 stats, pass1 emit) ---
// w1 is pre-swizzled in LDS into WMMA B-fragment order:
//   Wf[((jt*4+kk)*32 + lane)*16 + e] = w1[kk*32 + (lane&16)/2 + (e&7) + 2*(e&8)][16*jt + (lane&15)]
// so each B fragment is one 32B contiguous run per lane (2x ds_load_b128).
#define WF_HALVES_ (2 * C0_ * C1_)              // 16384
template <int PASS>
__global__ __launch_bounds__(256) void ec1_kernel(
    const _Float16* __restrict__ h0, const int* __restrict__ idx1,
    const float* __restrict__ w1, const float* __restrict__ b1,
    const float* __restrict__ g1, const float* __restrict__ beta1,
    float* __restrict__ stats1, float* __restrict__ act1,
    float* __restrict__ pooled) {
  extern __shared__ char smem[];
  _Float16* Hs   = (_Float16*)smem;                               // 18432 B
  _Float16* Wf   = (_Float16*)(smem + HS_BYTES_);                 // 32768 B
  float*    sred = (float*)(smem + HS_BYTES_ + 32768);            // 256 f
  float*    pacc = (float*)(smem + HS_BYTES_ + 32768 + 1024);     // 128 f
  const int b = blockIdx.x, tid = threadIdx.x;
  const _Float16* hb = h0 + (size_t)b * P_ * C0_;
  // async stage h0 tile (memory -> LDS, no VGPR bounce)
  for (int i = tid; i < 1024; i += 256) {
    const int row = i >> 3, ch = i & 7;
    async_ld_b128((unsigned)(row * (HP_ * 2) + ch * 16), hb + row * C0_ + ch * 8);
  }
  // swizzle w1 (f32 global) into f16 B-fragment order
  for (int i = tid; i < WF_HALVES_; i += 256) {
    const int jt = i >> 11, kk = (i >> 9) & 3, ln = (i >> 4) & 31, e = i & 15;
    const int f = kk * 32 + ((ln & 16) >> 1) + (e & 7) + ((e & 8) << 1);
    const int c = (jt << 4) + (ln & 15);
    Wf[i] = (_Float16)w1[f * C1_ + c];
  }
  if (PASS == 0) { if (tid < 256) sred[tid] = 0.f; }
  else           { if (tid < 128) pacc[tid] = 0.f; }
  wait_async0();
  __syncthreads();

  const int wave = tid >> 5, lane = tid & 31;
  const int hi = (lane >= 16) ? 1 : 0, e = lane & 15;
  const float invN = 1.f / (float)(B_ * P_ * K_);
  float bias[8], scl[8], shf[8];
#pragma unroll
  for (int jt = 0; jt < 8; ++jt) {
    const int c = 16 * jt + e;
    bias[jt] = b1[c];
    if (PASS == 1) {
      const float mean = stats1[c] * invN;
      const float var  = stats1[C1_ + c] * invN - mean * mean;
      const float s    = rsqrtf(var + EPS_) * g1[c];
      scl[jt] = s; shf[jt] = beta1[c] - mean * s;
    }
  }
  float sums[8], sqs[8];
#pragma unroll
  for (int j = 0; j < 8; ++j) { sums[j] = 0.f; sqs[j] = 0.f; }

  for (int p = wave; p < P_; p += 8) {       // 16 edge-tiles per wave (tile == point)
    const int nbr = idx1[((size_t)b * P_ + p) * K_ + e];
    const _Float16* rowi = &Hs[p * HP_];
    const _Float16* rowj = &Hs[nbr * HP_];
    v16h afr[4];
#pragma unroll
    for (int kk = 0; kk < 4; ++kk) {         // K steps of 32; f<64 = x_i, f>=64 = x_j
      const int ks = kk * 32;
      const _Float16* src = (ks < 64) ? rowi : rowj;
      const int base = (ks & 63) + hi * 8;
#pragma unroll
      for (int t = 0; t < 8; ++t) { afr[kk][t] = src[base + t]; afr[kk][8 + t] = src[base + 16 + t]; }
    }
#pragma unroll 1
    for (int jt = 0; jt < 8; ++jt) {
      v8f acc = {};
#pragma unroll
      for (int kk = 0; kk < 4; ++kk) {
        v16h bf;
        const _Float16* wp = &Wf[(((jt << 2) + kk) * 32 + lane) << 4];
#pragma unroll
        for (int t = 0; t < 16; ++t) bf[t] = wp[t];       // 32B contiguous
        acc = wmma_f16f32(afr[kk], bf, acc);
      }
      if (PASS == 0) {
#pragma unroll
        for (int r = 0; r < 8; ++r) {
          const float m = acc[r] + bias[jt];
          sums[jt] += m; sqs[jt] += m * m;
        }
      } else {
        const int c = 16 * jt + e;
        float* outp = act1 + ((size_t)b * P_ + p) * K_ * C1_ + c;
        float hsum = 0.f;
#pragma unroll
        for (int r = 0; r < 8; ++r) {
          const float m = acc[r] + bias[jt];
          float v = fmaxf(m * scl[jt] + shf[jt], 0.f);
          outp[(size_t)(r + hi * 8) * C1_] = v;   // edge row = r + 8*hi
          hsum += v;
        }
        hsum += __shfl_xor(hsum, 16, 32);         // join the two row halves
        if (!hi) atomicAdd(&pacc[c], hsum * (1.f / 16.f));  // mean over k
      }
    }
  }
  if (PASS == 0) {
#pragma unroll
    for (int jt = 0; jt < 8; ++jt) {
      const float s = sums[jt] + __shfl_xor(sums[jt], 16, 32);
      const float q = sqs[jt]  + __shfl_xor(sqs[jt], 16, 32);
      if (!hi) {
        const int c = 16 * jt + e;
        atomicAdd(&sred[c], s);
        atomicAdd(&sred[C1_ + c], q);
      }
    }
    __syncthreads();
    if (tid < 256) atomicAdd(&stats1[tid], sred[tid]);
  } else {
    __syncthreads();
    if (tid < 128) pooled[(size_t)b * C1_ + tid] = pacc[tid] * (1.f / (float)P_);  // mean over P
  }
}

// ---------------- kernel 7: global pool -> FC head ----------------------------
__global__ __launch_bounds__(256) void fc_kernel(
    const float* __restrict__ pooled, const float* __restrict__ fc1w,
    const float* __restrict__ fc1b, const float* __restrict__ fc2w,
    const float* __restrict__ fc2b, float* __restrict__ out) {
  __shared__ float pl[C1_];
  __shared__ float red[FC_];
  const int b = blockIdx.x, j = threadIdx.x;
  if (j < C1_) pl[j] = pooled[(size_t)b * C1_ + j];
  __syncthreads();
  float h = fc1b[j];
  for (int c = 0; c < C1_; ++c) h += pl[c] * fc1w[c * FC_ + j];
  red[j] = h * fc2w[j];
  __syncthreads();
  for (int st = 128; st > 0; st >>= 1) {
    if (j < st) red[j] += red[j + st];
    __syncthreads();
  }
  if (j == 0) out[b] = red[0] + fc2b[0];
}

// ---------------- host launcher ----------------------------------------------
extern "C" void kernel_launch(void* const* d_in, const int* in_sizes, int n_in,
                              void* d_out, int out_size, void* d_ws, size_t ws_size,
                              hipStream_t stream) {
  const float* x     = (const float*)d_in[0];
  const float* w0    = (const float*)d_in[1];
  const float* b0    = (const float*)d_in[2];
  const float* g0    = (const float*)d_in[3];
  const float* beta0 = (const float*)d_in[4];
  const float* w1    = (const float*)d_in[5];
  const float* b1    = (const float*)d_in[6];
  const float* g1    = (const float*)d_in[7];
  const float* beta1 = (const float*)d_in[8];
  const float* fc1w  = (const float*)d_in[9];
  const float* fc1b  = (const float*)d_in[10];
  const float* fc2w  = (const float*)d_in[11];
  const float* fc2b  = (const float*)d_in[12];

  float* out  = (float*)d_out;                                   // [512]
  float* act0 = out + B_;                                        // [512,128,16,64]
  float* act1 = act0 + (size_t)B_ * P_ * K_ * C0_;               // [512,128,16,128]

  char* ws = (char*)d_ws;
  int*      idx0   = (int*)ws;                                   // 4 MiB
  int*      idx1   = (int*)(ws + ((size_t)4 << 20));             // 4 MiB
  _Float16* h0f16  = (_Float16*)(ws + ((size_t)8 << 20));        // 8 MiB
  float*    stats0 = (float*)(ws + ((size_t)16 << 20));          // 128 f
  float*    stats1 = (float*)(ws + ((size_t)16 << 20) + 1024);   // 256 f
  float*    pooled = (float*)(ws + ((size_t)16 << 20) + 4096);   // 256 KiB

  init_stats_kernel<<<1, 384, 0, stream>>>(stats0, stats1);
  knn0_kernel<<<B_, 128, 0, stream>>>(x, idx0);
  ec0_pass1_kernel<<<B_, 256, 0, stream>>>(x, idx0, w0, b0, stats0);
  ec0_pass2_kernel<<<B_ * P_, 64, 0, stream>>>(x, idx0, w0, b0, g0, beta0, stats0,
                                               act0, h0f16);
  const size_t knn1_lds = HS_BYTES_ + 512 + (size_t)P_ * GP_ * 4;
  knn1_kernel<<<B_, 256, knn1_lds, stream>>>(h0f16, idx1);
  const size_t ec1_lds = HS_BYTES_ + 32768 + 1024 + 512;
  ec1_kernel<0><<<B_, 256, ec1_lds, stream>>>(h0f16, idx1, w1, b1, g1, beta1,
                                              stats1, act1, pooled);
  ec1_kernel<1><<<B_, 256, ec1_lds, stream>>>(h0f16, idx1, w1, b1, g1, beta1,
                                              stats1, act1, pooled);
  fc_kernel<<<B_, 256, 0, stream>>>(pooled, fc1w, fc1b, fc2w, fc2b, out);
}
